// HybridMemorySystem_41532333753042
// MI455X (gfx1250) — matrix operand order repaired and checked by reading, not compiled, not run on hardware
//
#include <hip/hip_runtime.h>
#include <hip/hip_bf16.h>
#include <math.h>

// ---------------- problem constants ----------------
constexpr int B_  = 8,  S_ = 512, I_ = 512, H_ = 512, L_ = 2, NH_ = 8;
constexpr int FF_ = 4 * H_;           // 2048
constexpr int HD_ = H_ / NH_;         // 64
constexpr int NMEM = 1024, WMEM = 64;
constexpr int DIF = NH_ * WMEM + NH_ + WMEM + 1 + WMEM + WMEM + 1;  // 714
constexpr int M_ROWS = B_ * S_;       // 4096

// ---------------- CDNA5 WMMA types ----------------
typedef __attribute__((ext_vector_type(16))) __bf16 v16bf;
typedef __attribute__((ext_vector_type(8)))  float  v8f;

// Native f32->bf16 convert (backend emits v_cvt_*bf16* on gfx1250; pairs get
// packed). Much cheaper than a manual RNE bit sequence on the WMMA feed path.
__device__ __forceinline__ __bf16 f2bf(float f) { return (__bf16)f; }

__device__ __forceinline__ v8f v8f_zero() {
    v8f z;
#pragma unroll
    for (int e = 0; e < 8; ++e) z[e] = 0.0f;
    return z;
}

__device__ __forceinline__ float softplus_(float x) { return (x > 20.f) ? x : log1pf(__expf(x)); }
__device__ __forceinline__ float sigmoid_(float x)  { return 1.f / (1.f + __expf(-x)); }

// A-fragment (16x32 bf16) direct from global fp32 row pointer (at column k0).
// Lane layout: lane = 16*khalf + row; elems 0..15 = K[16*khalf..16*khalf+15].
__device__ __forceinline__ v16bf load_afrag(const float* __restrict__ rowk0, int khalf, float scale) {
    v16bf a;
    const float* p0 = rowk0 + 16 * khalf;
    const float* p1 = rowk0 + 16 * khalf + 8;
#pragma unroll
    for (int j = 0; j < 8; ++j) {
        a[j]     = f2bf(p0[j] * scale);
        a[j + 8] = f2bf(p1[j] * scale);
    }
    return a;
}

// =====================================================================
// Generic tiled GEMM: C[M,N] = act(A[M,K] @ W[K,N] + bias[N] (+ R[M,N]))
// block = 128 threads (4 waves); WG tile 64 rows x 64 cols; wave tile 16x64.
// K stepped by 64: two 32-deep W chunks staged per barrier, pre-swizzled to
// B-fragment order so each lane reads its v16bf as one contiguous ds_load.
// Staging uses a loop-invariant column per thread -> fully unrolled,
// guard-hoisted, coalesced loads (no exec-mask churn).
// =====================================================================
__global__ void __launch_bounds__(128) gemm_bias_kernel(
    const float* __restrict__ A, const float* __restrict__ W,
    const float* __restrict__ bias, const float* __restrict__ R,
    float* __restrict__ C, int M, int N, int K, int relu)
{
    __shared__ __align__(32) __bf16 Bbuf[2][4][32][16];

    const int tid   = threadIdx.x;
    const int lane  = tid & 31;
    const int wid   = tid >> 5;
    const int lr    = lane & 15;
    const int khalf = lane >> 4;
    const int m0    = blockIdx.y * 64 + wid * 16;
    const int n0w   = blockIdx.x * 64;

    // staging geometry: column is loop-invariant per thread
    const int  sn    = tid & 63;          // tile column 0..63
    const int  kk0   = tid >> 6;          // starting k row (0 or 1)
    const int  scol  = n0w + sn;
    const bool svalid = (scol < N);
    const int  sct   = sn >> 4;
    const int  slr   = sn & 15;

    v8f acc[4];
#pragma unroll
    for (int ct = 0; ct < 4; ++ct) acc[ct] = v8f_zero();

    const float* arow = A + (size_t)(m0 + lr) * K;

    for (int k0 = 0; k0 < K; k0 += 64) {
        // stage W[k0..k0+63][n0w..n0w+63] -> B-fragment-ordered LDS (bf16)
        {
            const float* wp = W + (size_t)(k0 + kk0) * N + scol;
#pragma unroll
            for (int it = 0; it < 32; ++it) {
                int   kk = kk0 + 2 * it;              // 0..63
                float w  = svalid ? wp[(size_t)(2 * it) * N] : 0.0f;
                int   kc = kk >> 5;                   // which 32-deep chunk
                int   ki = kk & 31;
                int   tl = ((ki >> 4) << 4) + slr;
                Bbuf[kc][sct][tl][ki & 15] = f2bf(w);
            }
        }
        __syncthreads();

        __builtin_prefetch(arow + k0 + 64, 0, 1);   // global_prefetch on A stream
        v16bf af0 = load_afrag(arow + k0,      khalf, 1.0f);
        v16bf af1 = load_afrag(arow + k0 + 32, khalf, 1.0f);
#pragma unroll
        for (int ct = 0; ct < 4; ++ct) {
            v16bf bf0 = *reinterpret_cast<const v16bf*>(&Bbuf[0][ct][lane][0]);
            acc[ct] = __builtin_amdgcn_wmma_f32_16x16x32_bf16(
                false, af0, false, bf0, (short)0, acc[ct], false, false);
            v16bf bf1 = *reinterpret_cast<const v16bf*>(&Bbuf[1][ct][lane][0]);
            acc[ct] = __builtin_amdgcn_wmma_f32_16x16x32_bf16(
                false, af1, false, bf1, (short)0, acc[ct], false, false);
        }
        __syncthreads();
    }

#pragma unroll
    for (int ct = 0; ct < 4; ++ct) {
        int col = n0w + ct * 16 + lr;
        if (col < N) {
#pragma unroll
            for (int v = 0; v < 8; ++v) {
                int row = m0 + (khalf << 3) + v;
                float c = acc[ct][v] + bias[col];
                if (R) c += R[(size_t)row * N + col];
                if (relu) c = fmaxf(c, 0.0f);
                C[(size_t)row * N + col] = c;
            }
        }
    }
}

// =====================================================================
// Flash attention: one wave per (batch, head, 16-row query block).
// qkv layout: [B, S, 3H] with q|k|v concatenated; HD = 64.
// =====================================================================
__global__ void __launch_bounds__(32) flash_attn_kernel(
    const float* __restrict__ qkv, float* __restrict__ attnout)
{
    __shared__ __align__(32) __bf16 Pbuf[32][16];

    const int lane  = threadIdx.x & 31;
    const int lr    = lane & 15;
    const int khalf = lane >> 4;
    const int q0 = blockIdx.x * 16;
    const int h  = blockIdx.y;
    const int b  = blockIdx.z;

    const size_t rs3 = (size_t)3 * H_;
    const float* base  = qkv + (size_t)b * S_ * rs3;
    const float* qbase = base + h * HD_;
    const float* kbase = base + H_ + h * HD_;
    const float* vbase = base + 2 * H_ + h * HD_;

    // Q fragments for k-chunks {0..31},{32..63}; fold 1/sqrt(HD)=0.125 into Q
    v16bf qf[2];
    {
        const float* qrow = qbase + (size_t)(q0 + lr) * rs3;
#pragma unroll
        for (int kc = 0; kc < 2; ++kc)
            qf[kc] = load_afrag(qrow + kc * 32, khalf, 0.125f);
    }

    v8f o[4];
#pragma unroll
    for (int ct = 0; ct < 4; ++ct) o[ct] = v8f_zero();
    float mi[8], li[8];
#pragma unroll
    for (int v = 0; v < 8; ++v) { mi[v] = -3.0e38f; li[v] = 0.0f; }

    for (int j0 = 0; j0 < S_; j0 += 32) {
        // scores S(16x32) = Q(16x64) @ K_blk^T(64x32): 2 col-tiles x 2 k-chunks
        v8f s[2]; s[0] = v8f_zero(); s[1] = v8f_zero();
#pragma unroll
        for (int jt = 0; jt < 2; ++jt) {
#pragma unroll
            for (int kc = 0; kc < 2; ++kc) {
                const float* krow = kbase + (size_t)(j0 + jt * 16 + lr) * rs3
                                         + kc * 32 + 16 * khalf;
                v16bf kf;
#pragma unroll
                for (int j = 0; j < 16; ++j) kf[j] = f2bf(krow[j]);
                s[jt] = __builtin_amdgcn_wmma_f32_16x16x32_bf16(
                    false, qf[kc], false, kf, (short)0, s[jt], false, false);
            }
        }
        // online softmax (rows live in 16-lane groups; shfl masks 1,2,4,8)
        float alpha[8];
#pragma unroll
        for (int v = 0; v < 8; ++v) {
            float tm = fmaxf(s[0][v], s[1][v]);
#pragma unroll
            for (int m = 8; m >= 1; m >>= 1) tm = fmaxf(tm, __shfl_xor(tm, m, 32));
            float nm = fmaxf(mi[v], tm);
            alpha[v] = __expf(mi[v] - nm);
            float p0 = __expf(s[0][v] - nm);
            float p1 = __expf(s[1][v] - nm);
            s[0][v] = p0; s[1][v] = p1;
            float rsum = p0 + p1;
#pragma unroll
            for (int m = 8; m >= 1; m >>= 1) rsum += __shfl_xor(rsum, m, 32);
            li[v] = li[v] * alpha[v] + rsum;
            mi[v] = nm;
        }
#pragma unroll
        for (int ct = 0; ct < 4; ++ct)
#pragma unroll
            for (int v = 0; v < 8; ++v) o[ct][v] *= alpha[v];

        // re-lay P (C layout) into A-fragment layout through LDS
        __syncthreads();
#pragma unroll
        for (int v = 0; v < 8; ++v) {
            int r = v + 8 * khalf;
#pragma unroll
            for (int jt = 0; jt < 2; ++jt) {
                int c  = jt * 16 + lr;
                int tl = ((c >> 4) << 4) + r;
                int e  = c & 15;
                Pbuf[tl][e] = f2bf(s[jt][v]);
            }
        }
        __syncthreads();
        v16bf pf = *reinterpret_cast<const v16bf*>(&Pbuf[lane][0]);

        // O += P(16x32) @ V_blk(32x64)
#pragma unroll
        for (int ct = 0; ct < 4; ++ct) {
            v16bf vf;
            const float* vcol = vbase + 16 * ct + lr;
#pragma unroll
            for (int j = 0; j < 16; ++j)
                vf[j] = f2bf(vcol[(size_t)(j0 + 16 * khalf + j) * rs3]);
            o[ct] = __builtin_amdgcn_wmma_f32_16x16x32_bf16(
                false, pf, false, vf, (short)0, o[ct], false, false);
        }
    }

#pragma unroll
    for (int ct = 0; ct < 4; ++ct) {
        int col = h * HD_ + 16 * ct + lr;
#pragma unroll
        for (int v = 0; v < 8; ++v) {
            int row = q0 + v + 8 * khalf;
            attnout[((size_t)b * S_ + row) * H_ + col] = o[ct][v] / li[v];
        }
    }
}

// =====================================================================
// Row LayerNorm over H=512: one block of 256 threads per row.
// =====================================================================
__global__ void __launch_bounds__(256) ln_kernel(
    const float* __restrict__ X, const float* __restrict__ g,
    const float* __restrict__ bt, float* __restrict__ out)
{
    __shared__ float reds[8], reds2[8];
    const int row = blockIdx.x;
    const int tid = threadIdx.x;
    const float* x = X + (size_t)row * H_;
    float v0 = x[tid], v1 = x[tid + 256];
    float s = v0 + v1, s2 = v0 * v0 + v1 * v1;
#pragma unroll
    for (int m = 16; m >= 1; m >>= 1) { s += __shfl_xor(s, m, 32); s2 += __shfl_xor(s2, m, 32); }
    if ((tid & 31) == 0) { reds[tid >> 5] = s; reds2[tid >> 5] = s2; }
    __syncthreads();
    float ts = 0.f, ts2 = 0.f;
#pragma unroll
    for (int i = 0; i < 8; ++i) { ts += reds[i]; ts2 += reds2[i]; }
    float mean = ts * (1.0f / H_);
    float var  = ts2 * (1.0f / H_) - mean * mean;
    float rstd = rsqrtf(var + 1e-5f);
    out[(size_t)row * H_ + tid]       = (v0 - mean) * rstd * g[tid]       + bt[tid];
    out[(size_t)row * H_ + tid + 256] = (v1 - mean) * rstd * g[tid + 256] + bt[tid + 256];
}

// =====================================================================
// DNC memory scan: 1 workgroup / batch, M[1024x64] fp32 resident in the
// 320KB CDNA5 LDS for all 512 steps (saves ~2GB of HBM/L2 traffic).
// 1024 threads: thread n owns memory row n.
// =====================================================================
__device__ __forceinline__ float block_reduce_max_(float v, float* red) {
#pragma unroll
    for (int m = 16; m >= 1; m >>= 1) v = fmaxf(v, __shfl_xor(v, m, 32));
    if ((threadIdx.x & 31) == 0) red[threadIdx.x >> 5] = v;
    __syncthreads();
    float r = red[0];
#pragma unroll
    for (int i = 1; i < 32; ++i) r = fmaxf(r, red[i]);
    __syncthreads();
    return r;
}
__device__ __forceinline__ float block_reduce_sum_(float v, float* red) {
#pragma unroll
    for (int m = 16; m >= 1; m >>= 1) v += __shfl_xor(v, m, 32);
    if ((threadIdx.x & 31) == 0) red[threadIdx.x >> 5] = v;
    __syncthreads();
    float r = 0.f;
#pragma unroll
    for (int i = 0; i < 32; ++i) r += red[i];
    __syncthreads();
    return r;
}

__global__ void __launch_bounds__(1024) dnc_kernel(
    const float* __restrict__ iface, const float* __restrict__ M0,
    float* __restrict__ reads_out)
{
    extern __shared__ float smem[];
    float* Ms   = smem;                 // 65536 : M matrix
    float* sim  = Ms + NMEM * WMEM;     // 1024
    float* prob = sim + NMEM;           // 1024
    float* part = prob + NMEM;          // 1024 : 16 chunks x 64 cols
    float* ifr  = part + 1024;          // 768  : interface row
    float* red  = ifr + 768;            // 32
    float* rowi = red + 32;             // 1024 : inverse row norms

    const int tid = threadIdx.x;        // == memory row n
    const int b   = blockIdx.x;

    for (int i = tid; i < NMEM * WMEM; i += 1024) Ms[i] = M0[i];
    __syncthreads();

    for (int t = 0; t < S_; ++t) {
        const float* ifp = iface + ((size_t)b * S_ + t) * DIF;
        if (tid < DIF) ifr[tid] = ifp[tid];
        __syncthreads();

        // --- write addressing: ww = gw * softmax(wb * cos(M, wk)) ---
        float wkn = 0.f;
#pragma unroll 8
        for (int w = 0; w < WMEM; ++w) { float k = ifr[520 + w]; wkn += k * k; }
        wkn = rsqrtf(wkn + 1e-8f);
        const float wb = softplus_(ifr[584]);
        const float gw = sigmoid_(ifr[713]);
        {
            float rsq = 0.f, dot = 0.f;
            const float* mrow = Ms + tid * WMEM;
#pragma unroll 8
            for (int w = 0; w < WMEM; ++w) { float m = mrow[w]; rsq += m * m; dot += m * ifr[520 + w]; }
            sim[tid] = wb * dot * rsqrtf(rsq + 1e-8f) * wkn;
        }
        __syncthreads();
        float mx = block_reduce_max_(sim[tid], red);
        float ev = __expf(sim[tid] - mx);
        float sm = block_reduce_sum_(ev, red);
        float wwn = gw * ev / sm;

        // --- memory update (thread owns its row) + new row norm ---
        {
            float rsq = 0.f;
            float* mrow = Ms + tid * WMEM;
#pragma unroll 8
            for (int w = 0; w < WMEM; ++w) {
                float er = sigmoid_(ifr[585 + w]);
                float wv = ifr[649 + w];
                float m  = mrow[w];
                m = m * (1.f - wwn * er) + wwn * wv;
                mrow[w] = m;
                rsq += m * m;
            }
            rowi[tid] = rsqrtf(rsq + 1e-8f);
        }
        __syncthreads();

        // --- reads: per head, rw = softmax(rb * cos(M, rk)); r = rw @ M ---
        float* rout = reads_out + ((size_t)b * S_ + t) * (NH_ * WMEM);
        for (int hh = 0; hh < NH_; ++hh) {
            float rkn = 0.f;
#pragma unroll 8
            for (int w = 0; w < WMEM; ++w) { float k = ifr[hh * WMEM + w]; rkn += k * k; }
            rkn = rsqrtf(rkn + 1e-8f);
            const float rb = softplus_(ifr[512 + hh]);
            float dot = 0.f;
            const float* mrow = Ms + tid * WMEM;
#pragma unroll 8
            for (int w = 0; w < WMEM; ++w) dot += mrow[w] * ifr[hh * WMEM + w];
            float sv = rb * dot * rowi[tid] * rkn;
            float m2 = block_reduce_max_(sv, red);
            float e2 = __expf(sv - m2);
            float s2 = block_reduce_sum_(e2, red);
            prob[tid] = e2 / s2;
            __syncthreads();

            // r[w] = sum_n prob[n]*M[n][w] : 16 chunks of 64 rows
            {
                int w = tid & 63, c = tid >> 6;
                float acc = 0.f;
                int nb = c * 64;
#pragma unroll 8
                for (int i = 0; i < 64; ++i) acc += prob[nb + i] * Ms[(size_t)(nb + i) * WMEM + w];
                part[tid] = acc;
            }
            __syncthreads();
            if (tid < WMEM) {
                float r = 0.f;
#pragma unroll
                for (int c = 0; c < 16; ++c) r += part[c * 64 + tid];
                rout[hh * WMEM + tid] = r;
            }
            __syncthreads();
        }
        __syncthreads();
    }
}

// =====================================================================
// Orchestration
// =====================================================================
extern "C" void kernel_launch(void* const* d_in, const int* in_sizes, int n_in,
                              void* d_out, int out_size, void* d_ws, size_t ws_size,
                              hipStream_t stream) {
    (void)in_sizes; (void)n_in; (void)out_size; (void)ws_size;

    const float* x     = (const float*)d_in[0];
    const float* Wp    = (const float*)d_in[1];
    const float* bp    = (const float*)d_in[2];
    const float* Wqkv  = (const float*)d_in[3];
    const float* bqkv  = (const float*)d_in[4];
    const float* Wo    = (const float*)d_in[5];
    const float* bo    = (const float*)d_in[6];
    const float* ln1g  = (const float*)d_in[7];
    const float* ln1b  = (const float*)d_in[8];
    const float* W1    = (const float*)d_in[9];
    const float* b1    = (const float*)d_in[10];
    const float* W2    = (const float*)d_in[11];
    const float* b2    = (const float*)d_in[12];
    const float* ln2g  = (const float*)d_in[13];
    const float* ln2b  = (const float*)d_in[14];
    const float* Wi    = (const float*)d_in[15];
    const float* bi    = (const float*)d_in[16];
    const float* M0    = (const float*)d_in[17];

    float* out_t = (float*)d_out;                       // tout [B,S,H]
    float* out_r = out_t + (size_t)B_ * S_ * H_;        // reads [B,S,512]

    // workspace layout (floats)
    float* ws      = (float*)d_ws;
    float* hbuf    = ws;                                   // 4096*512
    float* qkvbuf  = hbuf    + (size_t)M_ROWS * H_;        // 4096*1536
    float* attnout = qkvbuf  + (size_t)M_ROWS * 3 * H_;    // 4096*512
    float* tmpbuf  = attnout + (size_t)M_ROWS * H_;        // 4096*512
    float* ffbuf   = tmpbuf  + (size_t)M_ROWS * H_;        // 4096*2048
    float* ifacebf = ffbuf   + (size_t)M_ROWS * FF_;       // 4096*714

    const dim3 gblk(128);

    // h = x @ Wp + bp
    gemm_bias_kernel<<<dim3(H_ / 64, M_ROWS / 64), gblk, 0, stream>>>(
        x, Wp, bp, nullptr, hbuf, M_ROWS, H_, I_, 0);

    for (int l = 0; l < L_; ++l) {
        const float* Wqkv_l = Wqkv + (size_t)l * H_ * 3 * H_;
        const float* bqkv_l = bqkv + (size_t)l * 3 * H_;
        gemm_bias_kernel<<<dim3(3 * H_ / 64, M_ROWS / 64), gblk, 0, stream>>>(
            hbuf, Wqkv_l, bqkv_l, nullptr, qkvbuf, M_ROWS, 3 * H_, H_, 0);

        flash_attn_kernel<<<dim3(S_ / 16, NH_, B_), 32, 0, stream>>>(qkvbuf, attnout);

        gemm_bias_kernel<<<dim3(H_ / 64, M_ROWS / 64), gblk, 0, stream>>>(
            attnout, Wo + (size_t)l * H_ * H_, bo + (size_t)l * H_, hbuf, tmpbuf,
            M_ROWS, H_, H_, 0);
        ln_kernel<<<M_ROWS, 256, 0, stream>>>(tmpbuf, ln1g + (size_t)l * H_, ln1b + (size_t)l * H_, hbuf);

        gemm_bias_kernel<<<dim3(FF_ / 64, M_ROWS / 64), gblk, 0, stream>>>(
            hbuf, W1 + (size_t)l * H_ * FF_, b1 + (size_t)l * FF_, nullptr, ffbuf,
            M_ROWS, FF_, H_, 1);
        gemm_bias_kernel<<<dim3(H_ / 64, M_ROWS / 64), gblk, 0, stream>>>(
            ffbuf, W2 + (size_t)l * FF_ * H_, b2 + (size_t)l * H_, hbuf, tmpbuf,
            M_ROWS, H_, FF_, 0);

        float* lnout = (l == L_ - 1) ? out_t : hbuf;     // final LN writes tout
        ln_kernel<<<M_ROWS, 256, 0, stream>>>(tmpbuf, ln2g + (size_t)l * H_, ln2b + (size_t)l * H_, lnout);
    }

    // iface = tout @ Wi + bi  (N = 714, guarded)
    gemm_bias_kernel<<<dim3((DIF + 63) / 64, M_ROWS / 64), gblk, 0, stream>>>(
        out_t, Wi, bi, nullptr, ifacebf, M_ROWS, DIF, H_, 0);

    // DNC scan: one WGP-resident workgroup per batch, M in LDS (~275 KB)
    const size_t dnc_smem =
        (size_t)(NMEM * WMEM + NMEM + NMEM + 1024 + 768 + 32 + NMEM) * sizeof(float);
    dnc_kernel<<<B_, 1024, dnc_smem, stream>>>(ifacebf, M0, out_r);
}
